// Video_Bridge_Block_60069412601979
// MI455X (gfx1250) — matrix operand order repaired and checked by reading, not compiled, not run on hardware
//
#include <hip/hip_runtime.h>

typedef __bf16 bf16;
typedef __attribute__((ext_vector_type(16))) __bf16 v16bf;
typedef __attribute__((ext_vector_type(8)))  __bf16 v8bf;
typedef __attribute__((ext_vector_type(8)))  float  v8f;
typedef __attribute__((ext_vector_type(4)))  unsigned int v4u;
typedef __attribute__((ext_vector_type(8)))  int v8i;
typedef __attribute__((ext_vector_type(4)))  int v4i;

// ---------- helpers ----------
__device__ __forceinline__ float bf2f(bf16 x) {
    union { unsigned short s; bf16 b; } in; in.b = x;
    union { unsigned u; float f; } out; out.u = ((unsigned)in.s) << 16;
    return out.f;
}
__device__ __forceinline__ bf16 f2bf(float f) {
    union { unsigned u; float f; } in; in.f = f;
    unsigned u = in.u + 0x7FFFu + ((in.u >> 16) & 1u);   // RNE
    union { unsigned short s; bf16 b; } out; out.s = (unsigned short)(u >> 16);
    return out.b;
}
__device__ __forceinline__ float waveSum(float v) {
    #pragma unroll
    for (int o = 16; o > 0; o >>= 1) v += __shfl_xor(v, o, 32);
    return v;
}
__device__ __forceinline__ float waveMax(float v) {
    #pragma unroll
    for (int o = 16; o > 0; o >>= 1) v = fmaxf(v, __shfl_xor(v, o, 32));
    return v;
}
__device__ __forceinline__ float blockSum(float v, float* red) {
    v = waveSum(v);
    int lane = threadIdx.x & 31, w = threadIdx.x >> 5;
    if (lane == 0) red[w] = v;
    __syncthreads();
    float t = 0.f;
    #pragma unroll
    for (int i = 0; i < 8; ++i) t += red[i];
    __syncthreads();
    return t;
}
__device__ __forceinline__ float blockMax(float v, float* red) {
    v = waveMax(v);
    int lane = threadIdx.x & 31, w = threadIdx.x >> 5;
    if (lane == 0) red[w] = v;
    __syncthreads();
    float t = -1e30f;
    #pragma unroll
    for (int i = 0; i < 8; ++i) t = fmaxf(t, red[i]);
    __syncthreads();
    return t;
}

// ---------- gate: p = LN(img_cls_emb) @ gate_W + gate_b  (16 x 768) ----------
__global__ __launch_bounds__(256) void gate_kernel(
    const float* __restrict__ emb, const float* __restrict__ g, const float* __restrict__ bb,
    const float* __restrict__ W, const float* __restrict__ bias, float* __restrict__ pOut)
{
    int b = blockIdx.x;
    __shared__ float xn[512];
    __shared__ float red[8];
    const float* x = emb + (long long)b * 512;
    float s = 0.f;
    for (int i = threadIdx.x; i < 512; i += 256) s += x[i];
    float mu = blockSum(s, red) / 512.f;
    float v = 0.f;
    for (int i = threadIdx.x; i < 512; i += 256) { float d = x[i] - mu; v += d * d; }
    float var = blockSum(v, red) / 512.f;
    float inv = rsqrtf(var + 1e-5f);
    for (int i = threadIdx.x; i < 512; i += 256) xn[i] = (x[i] - mu) * inv * g[i] + bb[i];
    __syncthreads();
    for (int c = threadIdx.x; c < 768; c += 256) {
        float acc = bias[c];
        for (int k = 0; k < 512; ++k) acc += xn[k] * W[(long long)k * 768 + c];
        pOut[(long long)b * 768 + c] = acc;
    }
}

// ---------- generic LayerNorm (f32 in -> bf16 out) ----------
__global__ __launch_bounds__(256) void ln_kernel(
    const float* __restrict__ in, bf16* __restrict__ out, int rows, int D,
    const float* __restrict__ g, const float* __restrict__ bb)
{
    int row = blockIdx.x;
    if (row >= rows) return;
    const float* x = in + (long long)row * D;
    __shared__ float red[8];
    float s = 0.f;
    for (int i = threadIdx.x; i < D; i += 256) s += x[i];
    float mu = blockSum(s, red) / (float)D;
    float v = 0.f;
    for (int i = threadIdx.x; i < D; i += 256) { float d = x[i] - mu; v += d * d; }
    float var = blockSum(v, red) / (float)D;
    float inv = rsqrtf(var + 1e-5f);
    bf16* o = out + (long long)row * D;
    for (int i = threadIdx.x; i < D; i += 256) o[i] = f2bf((x[i] - mu) * inv * g[i] + bb[i]);
}

// ---------- xg = x*sigmoid(p)+p then LN4 (skip cls token), bf16 out ----------
__global__ __launch_bounds__(256) void xgln_kernel(
    const float* __restrict__ x, const float* __restrict__ p, bf16* __restrict__ out,
    const float* __restrict__ g, const float* __restrict__ bb)
{
    int z = blockIdx.x;                 // 0 .. 16*1568-1
    int b = z / 1568, t = z % 1568;
    const float* xr = x + ((long long)(b * 1569 + 1 + t)) * 768;
    const float* pr = p + (long long)b * 768;
    __shared__ float sx[768];
    __shared__ float red[8];
    for (int i = threadIdx.x; i < 768; i += 256) {
        float pv = pr[i];
        float sig = 1.f / (1.f + __expf(-pv));
        sx[i] = xr[i] * sig + pv;
    }
    __syncthreads();
    float s = 0.f;
    for (int i = threadIdx.x; i < 768; i += 256) s += sx[i];
    float mu = blockSum(s, red) / 768.f;
    float v = 0.f;
    for (int i = threadIdx.x; i < 768; i += 256) { float d = sx[i] - mu; v += d * d; }
    float var = blockSum(v, red) / 768.f;
    float inv = rsqrtf(var + 1e-5f);
    bf16* o = out + (long long)z * 768;
    for (int i = threadIdx.x; i < 768; i += 256) o[i] = f2bf((sx[i] - mu) * inv * g[i] + bb[i]);
}

// ---------- WMMA GEMM: C = act(A_bf16[MxK] @ W_f32[KxN] + bias) (+resid) ----------
// BM=64, BN=128, BK=32; 8 waves, each wave owns a 32x32 output (4 WMMAs/k-step).
// A tile (64x32 bf16, row stride K) is fetched by the Tensor Data Mover
// (tensor_load_to_lds, TENSORcnt); rows past M read as zero per D# tensor_dim1.
#define BM 64
#define BN 128
#define BKK 32
__global__ __launch_bounds__(256) void gemm_wmma(
    const bf16* __restrict__ A, int M, int K,
    const float* __restrict__ W, int N,
    const float* __restrict__ bias,
    const float* __restrict__ resid,
    const int* __restrict__ resGate,
    void* __restrict__ outP, int outIsBF16, int actGelu)
{
    __shared__ __align__(16) bf16 sA[BM * BKK];     // [m][k]
    __shared__ __align__(16) bf16 sB[BN * BKK];     // [n][k] (transposed)
    int tid = threadIdx.x;
    int lane = tid & 31, wave = tid >> 5;
    int wm = wave >> 2, wn = wave & 3;              // 2 x 4 waves, 32x32 each
    long long m0 = (long long)blockIdx.y * BM;
    long long n0 = (long long)blockIdx.x * BN;
    v8f acc00 = {}, acc01 = {}, acc10 = {}, acc11 = {};
    int kt = K / BKK;
    int wr = tid >> 3, wc = (tid & 7) * 16;         // W: 32 rows x 128 cols, 16 floats/thread

    // --- TDM descriptor (wave 0 issues the async A-tile copy) ---
    unsigned ldsA_addr = (unsigned)(size_t)(&sA[0]);
    unsigned long long gbase = (unsigned long long)(size_t)(A + m0 * (long long)K);
    unsigned td0 = (unsigned)K;                                   // tensor_dim0 (elements)
    unsigned td1 = (unsigned)((M - m0) < BM ? (M - m0) : BM);     // tensor_dim1 (rows, OOB->0)
    unsigned long long st0 = (unsigned long long)(unsigned)K;     // dim0 stride (elements)
    v8i g1;
    g1[0] = (int)0x10000u;                                  // data_size=1 (2 bytes)
    g1[1] = (int)((td0 & 0xFFFFu) << 16);                   // tensor_dim0[15:0]
    g1[2] = (int)((td0 >> 16) | ((td1 & 0xFFFFu) << 16));   // dim0[31:16] | dim1[15:0]
    g1[3] = (int)((td1 >> 16) | ((unsigned)BKK << 16));     // dim1[31:16] | tile_dim0=32
    g1[4] = (int)(unsigned)BM;                              // tile_dim1=64, tile_dim2=0
    g1[5] = (int)(unsigned)(st0 & 0xFFFFFFFFull);           // dim0_stride[31:0]
    g1[6] = (int)(unsigned)((st0 >> 32) & 0xFFFFull);       // dim0_stride[47:32]
    g1[7] = 0;
    v4i gz4 = { 0, 0, 0, 0 };
    v8i gz8 = { 0, 0, 0, 0, 0, 0, 0, 0 };

    for (int t = 0; t < kt; ++t) {
        __syncthreads();
        if (wave == 0) {
            unsigned long long ga = gbase + (unsigned long long)t * (BKK * 2);
            v4u g0;
            g0[0] = 1u;                                          // count=1
            g0[1] = ldsA_addr;                                   // lds_addr
            g0[2] = (unsigned)(ga & 0xFFFFFFFFull);              // global_addr[31:0]
            g0[3] = (unsigned)((ga >> 32) & 0x01FFFFFFull) | (2u << 30);  // addr[56:32] | type=2
            __builtin_amdgcn_tensor_load_to_lds(g0, g1, gz4, gz4, gz8, 0);
            __builtin_amdgcn_s_wait_tensorcnt((short)0);
        }
        // W tile -> transposed bf16
        {
            const float* src = W + ((long long)(t * BKK + wr)) * N + n0 + wc;
            #pragma unroll
            for (int j = 0; j < 16; ++j) sB[(wc + j) * BKK + wr] = f2bf(src[j]);
        }
        __syncthreads();
        union { v16bf v; v8bf h8[2]; } af0, af1, bf0, bf1;
        {
            int kb = (lane < 16) ? 0 : 8;
            const bf16* ap0 = &sA[(wm * 32 + (lane & 15)) * BKK + kb];
            af0.h8[0] = *(const v8bf*)ap0;  af0.h8[1] = *(const v8bf*)(ap0 + 16);
            const bf16* ap1 = &sA[(wm * 32 + 16 + (lane & 15)) * BKK + kb];
            af1.h8[0] = *(const v8bf*)ap1;  af1.h8[1] = *(const v8bf*)(ap1 + 16);
        }
        {
            int kb = (lane < 16) ? 0 : 8;
            const bf16* bp0 = &sB[(wn * 32 + (lane & 15)) * BKK + kb];
            bf0.h8[0] = *(const v8bf*)bp0;  bf0.h8[1] = *(const v8bf*)(bp0 + 16);
            const bf16* bp1 = &sB[(wn * 32 + 16 + (lane & 15)) * BKK + kb];
            bf1.h8[0] = *(const v8bf*)bp1;  bf1.h8[1] = *(const v8bf*)(bp1 + 16);
        }
        acc00 = __builtin_amdgcn_wmma_f32_16x16x32_bf16(false, af0.v, false, bf0.v, (short)0, acc00, false, false);
        acc01 = __builtin_amdgcn_wmma_f32_16x16x32_bf16(false, af0.v, false, bf1.v, (short)0, acc01, false, false);
        acc10 = __builtin_amdgcn_wmma_f32_16x16x32_bf16(false, af1.v, false, bf0.v, (short)0, acc10, false, false);
        acc11 = __builtin_amdgcn_wmma_f32_16x16x32_bf16(false, af1.v, false, bf1.v, (short)0, acc11, false, false);
    }
    bool doRes = (resid != nullptr) && (resGate == nullptr || *resGate != 0);
    #pragma unroll
    for (int mi = 0; mi < 2; ++mi) {
        #pragma unroll
        for (int ni = 0; ni < 2; ++ni) {
            v8f acc = mi ? (ni ? acc11 : acc10) : (ni ? acc01 : acc00);
            long long n = n0 + wn * 32 + ni * 16 + (lane & 15);
            #pragma unroll
            for (int r = 0; r < 8; ++r) {
                long long m = m0 + wm * 32 + mi * 16 + r + ((lane < 16) ? 0 : 8);
                if (m < M) {
                    float v = acc[r];
                    if (bias) v += bias[n];
                    if (actGelu) v = 0.5f * v * (1.0f + erff(v * 0.70710678118654752f));
                    if (doRes) v += resid[m * N + n];
                    if (outIsBF16) ((bf16*)outP)[m * N + n] = f2bf(v);
                    else           ((float*)outP)[m * N + n] = v;
                }
            }
        }
    }
}

// ---------- WMMA flash-ish attention (16-query tile per wave) ----------
#define NKMAX 224
__global__ __launch_bounds__(32) void attn_wmma(
    const bf16* __restrict__ qP, long long qBS, int qRS, int qTok0, int qFS,
    const bf16* __restrict__ kvP, long long kBS, int kRS, int kTok0, int kFS,
    int kCO, int vCO, int prefix, int nq, int nkF,
    const float* __restrict__ maskP,
    bf16* __restrict__ outP, float scale)
{
    int z = blockIdx.y;
    int fr = z % 8, h = (z / 8) % 12, b = z / 96;
    int lane = threadIdx.x;
    int nk = prefix + nkF;
    int qi0 = blockIdx.x * 16;

    __shared__ __align__(16) bf16  sQ[16 * 64];
    __shared__ __align__(16) bf16  sKt[16 * 64];
    __shared__ __align__(16) bf16  sVt[64 * 32];
    __shared__ float sS[16 * NKMAX];
    __shared__ __align__(16) bf16  sP[16 * NKMAX];
    __shared__ float sSum[16];

    const bf16* qBase = qP + (long long)b * qBS + (long long)(qTok0 + fr * qFS) * qRS + h * 64;
    {
        int r = lane & 15, seg = lane >> 4;
        int qi = qi0 + r;
        bf16* dst = &sQ[r * 64 + seg * 32];
        if (qi < nq) {
            const bf16* src = qBase + (long long)qi * qRS + seg * 32;
            #pragma unroll
            for (int i = 0; i < 32; ++i) dst[i] = f2bf(bf2f(src[i]) * scale);
        } else {
            #pragma unroll
            for (int i = 0; i < 32; ++i) dst[i] = f2bf(0.f);
        }
    }
    __syncthreads();

    // pre-gather Q fragments (feature chunks 0..31 and 32..63)
    int arow = lane & 15;
    int akb  = (lane < 16) ? 0 : 8;
    int bkb  = (lane < 16) ? 0 : 16;
    union { v16bf v; v8bf h8[2]; } aq0, aq1;
    aq0.h8[0] = *(const v8bf*)&sQ[arow * 64 + akb];
    aq0.h8[1] = *(const v8bf*)&sQ[arow * 64 + 16 + akb];
    aq1.h8[0] = *(const v8bf*)&sQ[arow * 64 + 32 + akb];
    aq1.h8[1] = *(const v8bf*)&sQ[arow * 64 + 48 + akb];

    const bf16* kBase = kvP + (long long)b * kBS + (long long)(kTok0 + fr * kFS) * kRS + kCO + h * 64;
    const bf16* kPre  = kvP + (long long)b * kBS + kCO + h * 64;

    // S = Q @ K^T
    int nTiles = (nk + 15) >> 4;
    for (int ct = 0; ct < nTiles; ++ct) {
        __syncthreads();
        {
            int jr = lane & 15, seg = lane >> 4;
            int j = ct * 16 + jr;
            bf16* dst = &sKt[jr * 64 + seg * 32];
            if (j < nk) {
                const bf16* src = (prefix && j == 0) ? (kPre + seg * 32)
                                  : (kBase + (long long)(j - prefix) * kRS + seg * 32);
                #pragma unroll
                for (int i = 0; i < 32; i += 8) *(v8bf*)&dst[i] = *(const v8bf*)&src[i];
            } else {
                #pragma unroll
                for (int i = 0; i < 32; ++i) dst[i] = f2bf(0.f);
            }
        }
        __syncthreads();
        union { v16bf v; v8bf h8[2]; } bf0, bf1;
        int bn = lane & 15;
        bf0.h8[0] = *(const v8bf*)&sKt[bn * 64 + akb];
        bf0.h8[1] = *(const v8bf*)&sKt[bn * 64 + akb + 16];
        bf1.h8[0] = *(const v8bf*)&sKt[bn * 64 + 32 + akb];
        bf1.h8[1] = *(const v8bf*)&sKt[bn * 64 + 32 + akb + 16];
        v8f acc = {};
        acc = __builtin_amdgcn_wmma_f32_16x16x32_bf16(false, aq0.v, false, bf0.v, (short)0, acc, false, false);
        acc = __builtin_amdgcn_wmma_f32_16x16x32_bf16(false, aq1.v, false, bf1.v, (short)0, acc, false, false);
        #pragma unroll
        for (int r = 0; r < 8; ++r)
            sS[(r + ((lane < 16) ? 0 : 8)) * NKMAX + ct * 16 + (lane & 15)] = acc[r];
    }
    __syncthreads();

    // softmax over keys (f32), write P (bf16, zero-padded to multiple of 32)
    {
        int r = lane & 15;
        int start = (lane < 16) ? 0 : 1;
        const float* mrow = maskP ? (maskP + (long long)b * 77) : nullptr;
        float mx = -1e30f;
        for (int j = start; j < nk; j += 2) {
            float sv = sS[r * NKMAX + j];
            if (mrow) sv += (1.0f - mrow[j]) * -10000.0f;
            mx = fmaxf(mx, sv);
        }
        mx = fmaxf(mx, __shfl_xor(mx, 16, 32));
        int nkPad = ((nk + 31) / 32) * 32;
        float sm = 0.f;
        for (int j = start; j < nkPad; j += 2) {
            float pv = 0.f;
            if (j < nk) {
                float sv = sS[r * NKMAX + j];
                if (mrow) sv += (1.0f - mrow[j]) * -10000.0f;
                pv = __expf(sv - mx);
                sm += pv;
            }
            sP[r * NKMAX + j] = f2bf(pv);
        }
        sm += __shfl_xor(sm, 16, 32);
        if (lane < 16) sSum[r] = sm;
    }
    __syncthreads();

    // O = P @ V
    const bf16* vBase = kvP + (long long)b * kBS + (long long)(kTok0 + fr * kFS) * kRS + vCO + h * 64;
    const bf16* vPre  = kvP + (long long)b * kBS + vCO + h * 64;
    v8f o0 = {}, o1 = {}, o2 = {}, o3 = {};
    int nChunks = (nk + 31) >> 5;
    for (int c = 0; c < nChunks; ++c) {
        __syncthreads();
        {
            int j = c * 32 + lane;
            if (j < nk) {
                const bf16* src = (prefix && j == 0) ? vPre : (vBase + (long long)(j - prefix) * kRS);
                #pragma unroll
                for (int d2 = 0; d2 < 64; ++d2) sVt[d2 * 32 + lane] = src[d2];
            } else {
                #pragma unroll
                for (int d2 = 0; d2 < 64; ++d2) sVt[d2 * 32 + lane] = f2bf(0.f);
            }
        }
        __syncthreads();
        union { v16bf v; v8bf h8[2]; } ap;
        ap.h8[0] = *(const v8bf*)&sP[arow * NKMAX + c * 32 + akb];
        ap.h8[1] = *(const v8bf*)&sP[arow * NKMAX + c * 32 + 16 + akb];
        union { v16bf v; v8bf h8[2]; } bp0, bp1, bp2, bp3;
        int bn = lane & 15;
        bp0.h8[0] = *(const v8bf*)&sVt[(0 * 16 + bn) * 32 + akb];  bp0.h8[1] = *(const v8bf*)&sVt[(0 * 16 + bn) * 32 + akb + 16];
        bp1.h8[0] = *(const v8bf*)&sVt[(1 * 16 + bn) * 32 + akb];  bp1.h8[1] = *(const v8bf*)&sVt[(1 * 16 + bn) * 32 + akb + 16];
        bp2.h8[0] = *(const v8bf*)&sVt[(2 * 16 + bn) * 32 + akb];  bp2.h8[1] = *(const v8bf*)&sVt[(2 * 16 + bn) * 32 + akb + 16];
        bp3.h8[0] = *(const v8bf*)&sVt[(3 * 16 + bn) * 32 + akb];  bp3.h8[1] = *(const v8bf*)&sVt[(3 * 16 + bn) * 32 + akb + 16];
        o0 = __builtin_amdgcn_wmma_f32_16x16x32_bf16(false, ap.v, false, bp0.v, (short)0, o0, false, false);
        o1 = __builtin_amdgcn_wmma_f32_16x16x32_bf16(false, ap.v, false, bp1.v, (short)0, o1, false, false);
        o2 = __builtin_amdgcn_wmma_f32_16x16x32_bf16(false, ap.v, false, bp2.v, (short)0, o2, false, false);
        o3 = __builtin_amdgcn_wmma_f32_16x16x32_bf16(false, ap.v, false, bp3.v, (short)0, o3, false, false);
    }
    __syncthreads();
    // write out in "unheads" layout: token = 1 + fr*nq + qi, col = h*64 + d
    v8f oArr[4] = { o0, o1, o2, o3 };
    #pragma unroll
    for (int dt = 0; dt < 4; ++dt) {
        int ncol = dt * 16 + (lane & 15);
        #pragma unroll
        for (int r = 0; r < 8; ++r) {
            int m = r + ((lane < 16) ? 0 : 8);
            int qi = qi0 + m;
            if (qi < nq) {
                long long token = 1 + (long long)fr * nq + qi;
                float v = oArr[dt][r] / sSum[m];
                outP[((long long)b * 609 + token) * 768 + h * 64 + ncol] = f2bf(v);
            }
        }
    }
}

// ---------- single-query (cls) attention, VALU ----------
__global__ __launch_bounds__(256) void attn_cls(
    const bf16* __restrict__ qP, long long qBS,
    const bf16* __restrict__ kvP, long long kBS, int kRS, int kCO, int vCO,
    int nk, int maskMode, const float* __restrict__ maskP,
    bf16* __restrict__ outP, float scale)
{
    int z = blockIdx.x;
    int h = z % 12, b = z / 12;
    int tid = threadIdx.x;
    __shared__ float sq[64];
    __shared__ float sl[1600];
    __shared__ float red[8];
    __shared__ float oacc[4 * 64];

    const bf16* qr = qP + (long long)b * qBS + h * 64;
    if (tid < 64) sq[tid] = bf2f(qr[tid]) * scale;
    __syncthreads();

    const bf16* kb0 = kvP + (long long)b * kBS + kCO + h * 64;
    float mx = -1e30f;
    for (int j = tid; j < nk; j += 256) {
        const bf16* kr = kb0 + (long long)j * kRS;
        float acc = 0.f;
        #pragma unroll
        for (int d = 0; d < 64; ++d) acc += sq[d] * bf2f(kr[d]);
        if (maskMode) {
            int mc = (j == 0) ? 0 : (1 + ((j - 1) % 76));
            acc += (1.0f - maskP[(long long)b * 77 + mc]) * -10000.0f;
        }
        sl[j] = acc;
        mx = fmaxf(mx, acc);
    }
    mx = blockMax(mx, red);
    float sm = 0.f;
    for (int j = tid; j < nk; j += 256) {
        float p = __expf(sl[j] - mx);
        sl[j] = p;
        sm += p;
    }
    float total = blockSum(sm, red);
    __syncthreads();

    int d = tid & 63, part = tid >> 6;
    const bf16* vb0 = kvP + (long long)b * kBS + vCO + h * 64;
    float a = 0.f;
    for (int j = part; j < nk; j += 4) a += sl[j] * bf2f(vb0[(long long)j * kRS + d]);
    oacc[part * 64 + d] = a;
    __syncthreads();
    if (tid < 64) {
        float o = (oacc[tid] + oacc[64 + tid] + oacc[128 + tid] + oacc[192 + tid]) / total;
        outP[((long long)b * 609) * 768 + h * 64 + tid] = f2bf(o);
    }
}

// ---------- launch ----------
extern "C" void kernel_launch(void* const* d_in, const int* in_sizes, int n_in,
                              void* d_out, int out_size, void* d_ws, size_t ws_size,
                              hipStream_t stream)
{
    const float* x_bridge = (const float*)d_in[0];
    const float* x        = (const float*)d_in[1];
    const float* question = (const float*)d_in[2];
    const float* mask     = (const float*)d_in[3];
    const float* img      = (const float*)d_in[4];
    const float* ln4_g = (const float*)d_in[5],  *ln4_b = (const float*)d_in[6];
    const float* ln5_g = (const float*)d_in[7],  *ln5_b = (const float*)d_in[8];
    const float* ln6_g = (const float*)d_in[9],  *ln6_b = (const float*)d_in[10];
    const float* ln7_g = (const float*)d_in[11], *ln7_b = (const float*)d_in[12];
    const float* gln_g = (const float*)d_in[13], *gln_b = (const float*)d_in[14];
    const float* gate_W = (const float*)d_in[15], *gate_b = (const float*)d_in[16];
    const float* ca_Wq  = (const float*)d_in[17];
    const float* ca_Wkv = (const float*)d_in[18];
    const float* ca_Wp  = (const float*)d_in[19], *ca_bp = (const float*)d_in[20];
    const float* br_Wqkv= (const float*)d_in[21];
    const float* br_Wp  = (const float*)d_in[22], *br_bp = (const float*)d_in[23];
    const float* W1 = (const float*)d_in[24], *b1 = (const float*)d_in[25];
    const float* W2 = (const float*)d_in[26], *b2 = (const float*)d_in[27];
    const int* layer = (const int*)d_in[28];

    // geometry (matches setup_inputs)
    const int Mq = 16 * 77;          // 1232
    const int Mkv = 16 * 1568;       // 25088
    const int Mb = 16 * 609;         // 9744

    // workspace layout (bytes, lifetime-overlapped)
    char* ws = (char*)d_ws;
    float* p        = (float*)(ws + 0);                         //  16*768*4
    bf16*  lnq      = (bf16*)(ws + 49152);                      //  1232*512*2
    bf16*  qbuf     = (bf16*)(ws + 1310720);                    //  1232*768*2
    char*  slotC    = ws + 3203072;                             //  9744*768*2 (attnU/ln7/attnU2/ln6)
    char*  slotB    = ws + 18169856;                            //  max(lnkv bf16, bridge_temp f32)
    float* res      = (float*)(ws + 56705024);                  //  9744*768*4
    char*  slotA    = ws + 86638592;                            //  max(kvbuf, qkvbuf, hbuf)

    bf16*  lnkv        = (bf16*)slotB;
    bf16*  kvbuf       = (bf16*)slotA;
    bf16*  attnU       = (bf16*)slotC;
    float* bridge_temp = (float*)slotB;
    bf16*  ln7buf      = (bf16*)slotC;
    bf16*  qkvbuf      = (bf16*)slotA;
    bf16*  attnU2      = (bf16*)slotC;
    bf16*  ln6buf      = (bf16*)slotC;
    bf16*  hbuf        = (bf16*)slotA;

    const float scale = 0.125f;      // d=64 -> d^-0.5

    // 1. gate
    gate_kernel<<<16, 256, 0, stream>>>(img, gln_g, gln_b, gate_W, gate_b, p);
    // 2. ln5(question) -> lnq (bf16)
    ln_kernel<<<Mq, 256, 0, stream>>>(question, lnq, Mq, 512, ln5_g, ln5_b);
    // 3. q = lnq @ ca_Wq  -> qbuf (bf16)
    gemm_wmma<<<dim3(768 / 128, (Mq + 63) / 64), 256, 0, stream>>>(
        lnq, Mq, 512, ca_Wq, 768, nullptr, nullptr, nullptr, qbuf, 1, 0);
    // 4. xg + ln4 -> lnkv (bf16)
    xgln_kernel<<<Mkv, 256, 0, stream>>>(x, p, lnkv, ln4_g, ln4_b);
    // 5. kv = lnkv @ ca_Wkv -> kvbuf (bf16)
    gemm_wmma<<<dim3(1536 / 128, (Mkv + 63) / 64), 256, 0, stream>>>(
        lnkv, Mkv, 768, ca_Wkv, 1536, nullptr, nullptr, nullptr, kvbuf, 1, 0);
    // 6/7. cross attention (per-frame main tiles + cls query) -> attnU
    attn_wmma<<<dim3(5, 1536), 32, 0, stream>>>(
        qbuf, 77LL * 768, 768, 1, 0,
        kvbuf, 1568LL * 1536, 1536, 0, 196, 0, 768,
        0, 76, 196, nullptr, attnU, scale);
    attn_cls<<<192, 256, 0, stream>>>(
        qbuf, 77LL * 768, kvbuf, 1568LL * 1536, 1536, 0, 768,
        1568, 0, nullptr, attnU, scale);
    // 8. bridge_temp = attnU @ ca_Wproj + b (+ x_bridge if layer!=0)
    gemm_wmma<<<dim3(768 / 128, (Mb + 63) / 64), 256, 0, stream>>>(
        attnU, Mb, 768, ca_Wp, 768, ca_bp, x_bridge, layer, bridge_temp, 0, 0);
    // 9. ln7(bridge_temp) -> ln7buf
    ln_kernel<<<Mb, 256, 0, stream>>>(bridge_temp, ln7buf, Mb, 768, ln7_g, ln7_b);
    // 10. qkv = ln7buf @ br_Wqkv -> qkvbuf
    gemm_wmma<<<dim3(2304 / 128, (Mb + 63) / 64), 256, 0, stream>>>(
        ln7buf, Mb, 768, br_Wqkv, 2304, nullptr, nullptr, nullptr, qkvbuf, 1, 0);
    // 11/12. bridge attention (masked, cls-key prefix) -> attnU2
    attn_wmma<<<dim3(5, 1536), 32, 0, stream>>>(
        qkvbuf, 609LL * 2304, 2304, 1, 76,
        qkvbuf, 609LL * 2304, 2304, 1, 76, 768, 1536,
        1, 76, 76, mask, attnU2, scale);
    attn_cls<<<192, 256, 0, stream>>>(
        qkvbuf, 609LL * 2304, qkvbuf, 609LL * 2304, 2304, 768, 1536,
        609, 1, mask, attnU2, scale);
    // 13. res = attnU2 @ br_Wproj + b + bridge_temp
    gemm_wmma<<<dim3(768 / 128, (Mb + 63) / 64), 256, 0, stream>>>(
        attnU2, Mb, 768, br_Wp, 768, br_bp, bridge_temp, nullptr, res, 0, 0);
    // 14. ln6(res) -> ln6buf
    ln_kernel<<<Mb, 256, 0, stream>>>(res, ln6buf, Mb, 768, ln6_g, ln6_b);
    // 15. h = gelu(ln6buf @ W1 + b1) -> hbuf
    gemm_wmma<<<dim3(3072 / 128, (Mb + 63) / 64), 256, 0, stream>>>(
        ln6buf, Mb, 768, W1, 3072, b1, nullptr, nullptr, hbuf, 1, 1);
    // 16. out = res + h @ W2 + b2
    gemm_wmma<<<dim3(768 / 128, (Mb + 63) / 64), 256, 0, stream>>>(
        hbuf, Mb, 3072, W2, 768, b2, res, nullptr, d_out, 0, 0);
}
